// MoEMLP_35287451304344
// MI455X (gfx1250) — matrix operand order repaired and checked by reading, not compiled, not run on hardware
//
#include <hip/hip_runtime.h>
#include <math.h>
#include <stdint.h>

// ---------------------------------------------------------------------------
// Problem constants (from reference): B,S,D,H,E,TOP_K = 8,2048,1024,4096,8,2
// ---------------------------------------------------------------------------
#define BATCH 8
#define SEQ   2048
#define DIM   1024
#define HID   4096
#define NEXP  8

typedef __bf16 bf16;
typedef bf16  v16bf __attribute__((ext_vector_type(16)));
typedef bf16  v8bf  __attribute__((ext_vector_type(8)));
typedef bf16  v4bf  __attribute__((ext_vector_type(4)));
typedef float v8f   __attribute__((ext_vector_type(8)));
typedef unsigned int v4u __attribute__((ext_vector_type(4)));
typedef int  v4i __attribute__((ext_vector_type(4)));
typedef int  v8i __attribute__((ext_vector_type(8)));

// GEMM tiling
#define BM 128
#define BN 128
#define BK 32
#define LDSP 40   // padded LDS row (bf16): 80B rows, 16B aligned chunks

// Tensor Data Mover availability (device pass only; host pass uses fallback)
#if defined(__HIP_DEVICE_COMPILE__) && __has_builtin(__builtin_amdgcn_tensor_load_to_lds)
#define HAS_TDM 1
#else
#define HAS_TDM 0
#endif

// ---------------------------------------------------------------------------
// Gating: logits = x @ Wg + bg ; top-2 ; softmax over the 2.
// Token (0, t) for t < BATCH supplies the expert ids (bug-faithful reference).
// ---------------------------------------------------------------------------
__global__ __launch_bounds__(256)
void gate_kernel(const float* __restrict__ x, const float* __restrict__ Wg,
                 const float* __restrict__ bg, float* __restrict__ scores,
                 int* __restrict__ expert_ids) {
  int t = blockIdx.x * blockDim.x + threadIdx.x;   // token id in [0, B*S)
  if (t >= BATCH * SEQ) return;
  const float* xr = x + (size_t)t * DIM;

  float logit[NEXP];
#pragma unroll
  for (int e = 0; e < NEXP; ++e) logit[e] = bg[e];
  for (int d = 0; d < DIM; ++d) {
    float xv = xr[d];
#pragma unroll
    for (int e = 0; e < NEXP; ++e) logit[e] += xv * Wg[d * NEXP + e];
  }
  int i0 = 0; float v0 = logit[0];
#pragma unroll
  for (int e = 1; e < NEXP; ++e) if (logit[e] > v0) { v0 = logit[e]; i0 = e; }
  int i1 = -1; float v1 = -INFINITY;
#pragma unroll
  for (int e = 0; e < NEXP; ++e)
    if (e != i0 && logit[e] > v1) { v1 = logit[e]; i1 = e; }

  float e1 = __expf(v1 - v0);
  float inv = 1.0f / (1.0f + e1);
  scores[t * 2 + 0] = inv;
  scores[t * 2 + 1] = e1 * inv;

  if (t < BATCH) {
    expert_ids[0 * BATCH + t] = i0;
    expert_ids[1 * BATCH + t] = i1;
  }
}

// ---------------------------------------------------------------------------
// fp32 -> bf16 (row-major copy), n4 = elements/4
// ---------------------------------------------------------------------------
__global__ __launch_bounds__(256)
void cvt_kernel(const float* __restrict__ src, bf16* __restrict__ dst, size_t n4) {
  size_t i = (size_t)blockIdx.x * blockDim.x + threadIdx.x;
  if (i >= n4) return;
  float4 v = ((const float4*)src)[i];
  v4bf o;
  o[0] = (bf16)v.x; o[1] = (bf16)v.y; o[2] = (bf16)v.z; o[3] = (bf16)v.w;
  ((v4bf*)dst)[i] = o;
}

// ---------------------------------------------------------------------------
// fp32 [K][N] -> bf16 [N][K] transposed copy (per expert, tiled via LDS).
// grid: (N/32, K/32, E), block: 256 = 32x8
// ---------------------------------------------------------------------------
__global__ __launch_bounds__(256)
void transpose_cvt_kernel(const float* __restrict__ src, bf16* __restrict__ dst,
                          int K, int N) {
  __shared__ float tile[32][33];
  const size_t eoff = (size_t)blockIdx.z * K * N;
  src += eoff; dst += eoff;
  const int n0 = blockIdx.x * 32, k0 = blockIdx.y * 32;
  const int tx = threadIdx.x & 31, ty = threadIdx.x >> 5;
#pragma unroll
  for (int r = ty; r < 32; r += 8)
    tile[r][tx] = src[(size_t)(k0 + r) * N + (n0 + tx)];
  __syncthreads();
#pragma unroll
  for (int r = ty; r < 32; r += 8)
    dst[(size_t)(n0 + r) * K + (k0 + tx)] = (bf16)tile[tx][r];
}

// ---------------------------------------------------------------------------
// TDM: issue a 2D tile load (tile_d0 x tile_d1, bf16) global -> LDS, with
// LDS row padding 32 bf16 (64B = 16 DW) -> +8 bf16 (16B = 4 DW) per row.
// ---------------------------------------------------------------------------
#if HAS_TDM
__device__ __forceinline__ void tdm_load_tile(uint32_t lds_addr, const bf16* gptr,
                                              int tensor_d0, int tensor_d1,
                                              long long stride0) {
  const uint64_t ga = (uint64_t)(uintptr_t)gptr;
  v4u g0;
  g0[0] = 1u;                                   // count=1 (valid), user mode
  g0[1] = lds_addr;                             // LDS byte address
  g0[2] = (uint32_t)ga;                         // global_addr[31:0]
  g0[3] = (uint32_t)((ga >> 32) & 0x1FFFFFFu)   // global_addr[56:32]
        | (2u << 30);                           // type=2 ("image")
  v8i g1;
  g1[0] = (1 << 16)    // data_size = 1 -> 2 bytes (bf16)
        | (1 << 20)    // pad_enable
        | (3 << 22)    // pad_interval: 16 DWORDs (=64B = one 32-bf16 row)
        | (3 << 25);   // pad_amount: 4 DWORDs (=16B = 8 bf16) -> LDSP=40
  g1[1] = (int)(((uint32_t)tensor_d0 & 0xFFFFu) << 16);              // dim0[15:0]
  g1[2] = (int)(((uint32_t)tensor_d0 >> 16)
        | (((uint32_t)tensor_d1 & 0xFFFFu) << 16));                  // dim0 hi, dim1 lo
  g1[3] = (int)(((uint32_t)tensor_d1 >> 16) | ((uint32_t)BK << 16)); // dim1 hi, tile_d0=32
  g1[4] = BM;                                                        // tile_d1=128
  g1[5] = (int)(uint32_t)stride0;                                    // dim0_stride[31:0]
  g1[6] = (int)(uint32_t)((uint64_t)stride0 >> 32);                  // dim0_stride[47:32]
  g1[7] = 0;
  v4i z4 = {0, 0, 0, 0};
#if __clang_major__ >= 23
  v8i z8 = {0, 0, 0, 0, 0, 0, 0, 0};
  __builtin_amdgcn_tensor_load_to_lds(g0, g1, z4, z4, z8, 0);
#else
  __builtin_amdgcn_tensor_load_to_lds(g0, g1, z4, z4, 0);
#endif
}
#endif

// ---------------------------------------------------------------------------
// WMMA GEMM: C[M,N] = A[M,K] @ Bt[N,K]^T (bf16 in, f32 accumulate)
//   A   row-major [M][K]; Bt row-major [N][K] (weights pre-transposed)
//   MODE 0: epilogue = bias + exact GELU, store bf16 to hout
//   MODE 1: epilogue = (bias + acc) * gate_score; islot==0 overwrite, else +=
// Block: 256 threads = 8 waves; wave w owns 64x32 at ((w&1)*64,(w>>1)*32).
// TDM path: double-buffered LDS, wave 0 issues next tile pair while all
// waves compute the current one; s_wait_tensorcnt gates consumption.
// ---------------------------------------------------------------------------
template <int MODE>
__global__ __launch_bounds__(256)
void wmma_gemm_kernel(const bf16* __restrict__ A,
                      const bf16* __restrict__ Wbase, size_t expert_stride,
                      const int* __restrict__ expert_ids, int pair,
                      const float* __restrict__ bias_base,
                      bf16* __restrict__ hout,
                      float* __restrict__ out,
                      const float* __restrict__ scores,
                      int M, int N, int Kdim, int bidx, int islot) {
#if HAS_TDM
  __shared__ bf16 As[2][BM][LDSP];
  __shared__ bf16 Bs[2][BN][LDSP];
#else
  __shared__ bf16 As[1][BM][LDSP];
  __shared__ bf16 Bs[1][BN][LDSP];
#endif

  const int e = expert_ids[pair];                    // uniform load
  const bf16*  Bt   = Wbase + (size_t)e * expert_stride;  // [N][K]
  const float* bias = bias_base + (size_t)e * N;

  const int tid  = threadIdx.x;
  const int lane = tid & 31;
  const int wave = tid >> 5;
  const int lr = lane & 15;       // fragment row/col within 16
  const int lh = lane >> 4;       // K-half select per ISA layout
  const int wm = (wave & 1) * 64;
  const int wn = (wave >> 1) * 32;
  const int m0 = blockIdx.y * BM;
  const int n0 = blockIdx.x * BN;

  v8f acc[4][2];
#pragma unroll
  for (int mt = 0; mt < 4; ++mt)
#pragma unroll
    for (int nt = 0; nt < 2; ++nt)
#pragma unroll
      for (int j = 0; j < 8; ++j) acc[mt][nt][j] = 0.0f;

  const int nit = Kdim / BK;

#if HAS_TDM
  const uint32_t ldsA0 = (uint32_t)(uintptr_t)&As[0][0][0];
  const uint32_t ldsA1 = (uint32_t)(uintptr_t)&As[1][0][0];
  const uint32_t ldsB0 = (uint32_t)(uintptr_t)&Bs[0][0][0];
  const uint32_t ldsB1 = (uint32_t)(uintptr_t)&Bs[1][0][0];
  if (wave == 0) {   // prologue: tile 0 into buffer 0
    tdm_load_tile(ldsA0, A  + (size_t)m0 * Kdim, Kdim, M, Kdim);
    tdm_load_tile(ldsB0, Bt + (size_t)n0 * Kdim, Kdim, N, Kdim);
  }
#endif

  for (int it = 0; it < nit; ++it) {
    const int k0 = it * BK;
#if HAS_TDM
    const int buf = it & 1;
    if (wave == 0) {
      if (it + 1 < nit) {
        const int k1 = k0 + BK;
        tdm_load_tile(buf ? ldsA0 : ldsA1, A  + (size_t)m0 * Kdim + k1, Kdim, M, Kdim);
        tdm_load_tile(buf ? ldsB0 : ldsB1, Bt + (size_t)n0 * Kdim + k1, Kdim, N, Kdim);
        __builtin_amdgcn_s_wait_tensorcnt(2);   // current tile pair retired
      } else {
        __builtin_amdgcn_s_wait_tensorcnt(0);
      }
    }
    __syncthreads();   // tile `buf` visible to all waves
#else
    const int buf = 0;
    {  // manual staging: both tiles are contiguous [row][32] copies
      const int r = tid >> 1, c = (tid & 1) * 16;
      const bf16* sa = A  + (size_t)(m0 + r) * Kdim + (k0 + c);
      const bf16* sb = Bt + (size_t)(n0 + r) * Kdim + (k0 + c);
      v8bf a0 = *(const v8bf*)(sa);
      v8bf a1 = *(const v8bf*)(sa + 8);
      v8bf b0 = *(const v8bf*)(sb);
      v8bf b1 = *(const v8bf*)(sb + 8);
      *(v8bf*)&As[0][r][c]     = a0;
      *(v8bf*)&As[0][r][c + 8] = a1;
      *(v8bf*)&Bs[0][r][c]     = b0;
      *(v8bf*)&Bs[0][r][c + 8] = b1;
      if (k0 + BK < Kdim) {
        __builtin_prefetch(sa + BK, 0, 1);
        __builtin_prefetch(sb + BK, 0, 1);
      }
    }
    __syncthreads();
#endif

    // A 16x32 bf16 frag: lanes0-15 hold K 0..7 & 16..23; lanes16-31 K 8..15 & 24..31
    v16bf afrag[4];
#pragma unroll
    for (int mt = 0; mt < 4; ++mt) {
      const int row = wm + mt * 16 + lr;
      v8bf lo = *(const v8bf*)&As[buf][row][lh * 8];
      v8bf hi = *(const v8bf*)&As[buf][row][16 + lh * 8];
#pragma unroll
      for (int j = 0; j < 8; ++j) { afrag[mt][j] = lo[j]; afrag[mt][8 + j] = hi[j]; }
    }
    // B 32x16 frag: lane = column; lanes0-15 hold K 0..7 & 16..23; lanes16-31 K 8..15 & 24..31
    v16bf bfrag[2];
#pragma unroll
    for (int nt = 0; nt < 2; ++nt) {
      const int col = wn + nt * 16 + lr;
      v8bf lo = *(const v8bf*)&Bs[buf][col][lh * 8];
      v8bf hi = *(const v8bf*)&Bs[buf][col][16 + lh * 8];
#pragma unroll
      for (int j = 0; j < 8; ++j) { bfrag[nt][j] = lo[j]; bfrag[nt][8 + j] = hi[j]; }
    }

#pragma unroll
    for (int mt = 0; mt < 4; ++mt)
#pragma unroll
      for (int nt = 0; nt < 2; ++nt)
        acc[mt][nt] = __builtin_amdgcn_wmma_f32_16x16x32_bf16(
            false, afrag[mt], false, bfrag[nt],
            (short)0, acc[mt][nt], false, false);

    __syncthreads();   // all reads of `buf` done before it is overwritten
  }

  // ---- epilogue: C frag VGPR j -> M = j (lanes0-15) / 8+j (lanes16-31), N = lane&15
#pragma unroll
  for (int mt = 0; mt < 4; ++mt) {
#pragma unroll
    for (int nt = 0; nt < 2; ++nt) {
      const int colg = n0 + wn + nt * 16 + lr;
      const float bval = bias[colg];
#pragma unroll
      for (int j = 0; j < 8; ++j) {
        const int rowg = m0 + wm + mt * 16 + (lh ? (8 + j) : j);
        float v = acc[mt][nt][j] + bval;
        if (MODE == 0) {
          float g = 0.5f * v * (1.0f + erff(v * 0.70710678118654752f));
          hout[(size_t)rowg * N + colg] = (bf16)g;
        } else {
          const float s = scores[((size_t)bidx * SEQ + rowg) * 2 + islot];
          float val = s * v;
          float* op = out + ((size_t)bidx * SEQ + rowg) * N + colg;
          if (islot == 0) *op = val;
          else            *op += val;
        }
      }
    }
  }
}

// ---------------------------------------------------------------------------
// Host-side orchestration
// ---------------------------------------------------------------------------
extern "C" void kernel_launch(void* const* d_in, const int* in_sizes, int n_in,
                              void* d_out, int out_size, void* d_ws, size_t ws_size,
                              hipStream_t stream) {
  const float* x  = (const float*)d_in[0];
  const float* Wg = (const float*)d_in[1];
  const float* bg = (const float*)d_in[2];
  const float* W1 = (const float*)d_in[3];
  const float* b1 = (const float*)d_in[4];
  const float* W2 = (const float*)d_in[5];
  const float* b2 = (const float*)d_in[6];
  float* out = (float*)d_out;

  char* ws = (char*)d_ws;
  size_t off = 0;
  auto carve = [&](size_t bytes) -> char* {
    char* p = ws + off;
    off += (bytes + 255) & ~(size_t)255;
    return p;
  };
  int*   expert_ids = (int*)  carve(2 * BATCH * sizeof(int));
  float* scores     = (float*)carve((size_t)BATCH * SEQ * 2 * sizeof(float));
  bf16*  xbf        = (bf16*) carve((size_t)BATCH * SEQ * DIM * sizeof(bf16));
  bf16*  W1t        = (bf16*) carve((size_t)NEXP * DIM * HID * sizeof(bf16)); // [E][H][D]
  bf16*  W2t        = (bf16*) carve((size_t)NEXP * HID * DIM * sizeof(bf16)); // [E][D][H]
  bf16*  hbuf       = (bf16*) carve((size_t)SEQ * HID * sizeof(bf16));

  // 1) gating (scores + data-dependent expert ids)
  gate_kernel<<<(BATCH * SEQ + 255) / 256, 256, 0, stream>>>(x, Wg, bg, scores,
                                                             expert_ids);
  // 2) bf16 conversion: x row-major; weights transposed to [N][K] per expert
  {
    size_t nx4 = (size_t)BATCH * SEQ * DIM / 4;
    cvt_kernel<<<(unsigned)((nx4 + 255) / 256), 256, 0, stream>>>(x, xbf, nx4);
    dim3 tb(256);
    dim3 tg1(HID / 32, DIM / 32, NEXP);   // W1 [D][H] -> [H][D]
    transpose_cvt_kernel<<<tg1, tb, 0, stream>>>(W1, W1t, DIM, HID);
    dim3 tg2(DIM / 32, HID / 32, NEXP);   // W2 [H][D] -> [D][H]
    transpose_cvt_kernel<<<tg2, tb, 0, stream>>>(W2, W2t, HID, DIM);
  }

  // 3) expert passes; i=0 overwrites out, i=1 accumulates (stream-ordered,
  //    so the single hbuf is reused across the 16 pairs)
  dim3 blk(256);
  dim3 g1(HID / BN, SEQ / BM);   // 32 x 16 blocks
  dim3 g2(DIM / BN, SEQ / BM);   //  8 x 16 blocks
  for (int i = 0; i < 2; ++i) {
    for (int b = 0; b < BATCH; ++b) {
      const int pair = i * BATCH + b;
      wmma_gemm_kernel<0><<<g1, blk, 0, stream>>>(
          xbf + (size_t)b * SEQ * DIM, W1t, (size_t)DIM * HID,
          expert_ids, pair, b1, hbuf, nullptr, nullptr,
          SEQ, HID, DIM, b, i);
      wmma_gemm_kernel<1><<<g2, blk, 0, stream>>>(
          hbuf, W2t, (size_t)HID * DIM,
          expert_ids, pair, b2, nullptr, out, scores,
          SEQ, DIM, HID, b, i);
    }
  }
}
